// GATGraphNet_9259949490750
// MI455X (gfx1250) — matrix operand order repaired and checked by reading, not compiled, not run on hardware
//
#include <hip/hip_runtime.h>
#include <math.h>

// ---------------------------------------------------------------------------
// GAT forward (2x GATConv + linear) for MI455X (gfx1250, wave32).
//  - Dense feature transforms via V_WMMA_F32_16X16X32_F16 (fp16 in, fp32 acc)
//  - CSR-by-dst built once (count / LDS block-scan / scatter), then
//    per-node wave-parallel segment softmax with ZERO float atomics:
//    layer-2 cost = 850k coalesced 256B row gathers ~= 218 MB => ~10us @ 23.3TB/s.
//    (The old 3-pass atomic design issued ~54M f32 L2 atomics -- wrong path.)
// ---------------------------------------------------------------------------

typedef __attribute__((ext_vector_type(16))) _Float16 v16h;
typedef __attribute__((ext_vector_type(8)))  float    v8f;

constexpr int N_NODES   = 50000;
constexpr int NE        = 800000;          // directed edges (before self loops)
constexpr int NET       = NE + N_NODES;    // + self loops
constexpr int ROW_TILES = N_NODES / 16;    // 3125 exactly

// ======================= CSR construction ==================================

__global__ void k_zero_int(int* __restrict__ p, int n) {
    int i = blockIdx.x * blockDim.x + threadIdx.x;
    if (i < n) p[i] = 0;
}

__global__ void k_count(const int* __restrict__ dst, int* __restrict__ cnt) {
    int e = blockIdx.x * blockDim.x + threadIdx.x;
    if (e >= NET) return;
    int d = (e < NE) ? dst[e] : (e - NE);
    atomicAdd(&cnt[d], 1);
}

// single-workgroup exclusive scan over N_NODES counts -> rowptr[N+1]
__global__ void k_scan(const int* __restrict__ cnt, int* __restrict__ rowptr) {
    __shared__ int sm[1024];
    __shared__ int carry;
    int t = threadIdx.x;
    if (t == 0) carry = 0;
    __syncthreads();
    for (int base = 0; base < N_NODES; base += 1024) {
        int i = base + t;
        int v = (i < N_NODES) ? cnt[i] : 0;
        sm[t] = v;
        __syncthreads();
        for (int off = 1; off < 1024; off <<= 1) {   // Hillis-Steele inclusive
            int add = (t >= off) ? sm[t - off] : 0;
            __syncthreads();
            sm[t] += add;
            __syncthreads();
        }
        if (i < N_NODES) rowptr[i] = carry + sm[t] - v;  // exclusive
        __syncthreads();
        if (t == 1023) carry += sm[1023];
        __syncthreads();
    }
    if (t == 0) rowptr[N_NODES] = carry;
}

__global__ void k_copy_int(const int* __restrict__ a, int* __restrict__ b, int n) {
    int i = blockIdx.x * blockDim.x + threadIdx.x;
    if (i < n) b[i] = a[i];
}

__global__ void k_scatter(const int* __restrict__ src, const int* __restrict__ dst,
                          int* __restrict__ cursor, int* __restrict__ colidx) {
    int e = blockIdx.x * blockDim.x + threadIdx.x;
    if (e >= NET) return;
    int s = (e < NE) ? src[e] : (e - NE);
    int d = (e < NE) ? dst[e] : (e - NE);
    int pos = atomicAdd(&cursor[d], 1);
    colidx[pos] = s;
}

// ======================= WMMA GEMMs ========================================

// h1[N,16] = x[N,11] @ W1[11,16]
__global__ void k_gemm1(const float* __restrict__ x, const float* __restrict__ W,
                        float* __restrict__ h) {
    int wave = blockIdx.x * (blockDim.x >> 5) + (threadIdx.x >> 5);
    if (wave >= ROW_TILES) return;
    int lane = threadIdx.x & 31;
    int half = lane >> 4;
    int idx  = lane & 15;
    int row  = wave * 16 + idx;

    v16h a;
#pragma unroll
    for (int j = 0; j < 16; ++j) {
        int K = half * 16 + j;
        a[j] = (K < 11) ? (_Float16)x[row * 11 + K] : (_Float16)0.0f;
    }
    v16h b;
    int col = idx, kb = half * 16;
#pragma unroll
    for (int j = 0; j < 16; ++j) {
        int K = kb + j;
        b[j] = (K < 11) ? (_Float16)W[K * 16 + col] : (_Float16)0.0f;
    }
    v8f c = {};
    c = __builtin_amdgcn_wmma_f32_16x16x32_f16(false, a, false, b, (short)0, c, false, false);
#pragma unroll
    for (int i = 0; i < 8; ++i) {
        int m = i + half * 8;
        h[(wave * 16 + m) * 16 + col] = c[i];
    }
}

// h2[N,64] = relu(agg1[N,16]+b1) @ W2[16,64]   (bias + relu fused into A-load)
__global__ void k_gemm2(const float* __restrict__ agg1, const float* __restrict__ b1,
                        const float* __restrict__ W2, float* __restrict__ h2) {
    int wave = blockIdx.x * (blockDim.x >> 5) + (threadIdx.x >> 5);
    if (wave >= ROW_TILES) return;
    int ct   = blockIdx.y;
    int lane = threadIdx.x & 31;
    int half = lane >> 4;
    int idx  = lane & 15;
    int row  = wave * 16 + idx;

    v16h a;
#pragma unroll
    for (int j = 0; j < 16; ++j) {
        int K = half * 16 + j;
        float v = 0.0f;
        if (K < 16) { v = agg1[row * 16 + K] + b1[K]; v = v > 0.0f ? v : 0.0f; }
        a[j] = (_Float16)v;
    }
    v16h b;
    int col = idx, kb = half * 16;
#pragma unroll
    for (int j = 0; j < 16; ++j) {
        int K = kb + j;
        b[j] = (K < 16) ? (_Float16)W2[K * 64 + ct * 16 + col] : (_Float16)0.0f;
    }
    v8f c = {};
    c = __builtin_amdgcn_wmma_f32_16x16x32_f16(false, a, false, b, (short)0, c, false, false);
#pragma unroll
    for (int i = 0; i < 8; ++i) {
        int m = i + half * 8;
        h2[(wave * 16 + m) * 64 + ct * 16 + col] = c[i];
    }
}

// out[N,10] = (agg2[N,64]+b2) @ Wl[64,10] + bl   (both biases fused)
__global__ void k_gemm3(const float* __restrict__ agg2, const float* __restrict__ b2,
                        const float* __restrict__ Wl, const float* __restrict__ bl,
                        float* __restrict__ out) {
    int wave = blockIdx.x * (blockDim.x >> 5) + (threadIdx.x >> 5);
    if (wave >= ROW_TILES) return;
    int lane = threadIdx.x & 31;
    int half = lane >> 4;
    int idx  = lane & 15;
    int row  = wave * 16 + idx;
    int col  = idx;

    v8f c = {};
#pragma unroll
    for (int q = 0; q < 2; ++q) {           // K = 64 -> two 32-deep WMMAs
        v16h a;
#pragma unroll
        for (int j = 0; j < 16; ++j) {
            int K = q * 32 + half * 16 + j;
            a[j] = (_Float16)(agg2[row * 64 + K] + b2[K]);
        }
        v16h b;
        int kb = q * 32 + half * 16;
#pragma unroll
        for (int j = 0; j < 16; ++j) {
            int K = kb + j;
            b[j] = (col < 10) ? (_Float16)Wl[K * 10 + col] : (_Float16)0.0f;
        }
        c = __builtin_amdgcn_wmma_f32_16x16x32_f16(false, a, false, b, (short)0, c, false, false);
    }
    if (col < 10) {
#pragma unroll
        for (int i = 0; i < 8; ++i) {
            int m = i + half * 8;
            out[(wave * 16 + m) * 10 + col] = c[i] + bl[col];
        }
    }
}

// ======================= attention =========================================

template <int C>
__global__ void k_alpha(const float* __restrict__ h, const float* __restrict__ aw_s,
                        const float* __restrict__ aw_d, float* __restrict__ as,
                        float* __restrict__ ad) {
    int i = blockIdx.x * blockDim.x + threadIdx.x;
    if (i >= N_NODES) return;
    float s = 0.0f, d = 0.0f;
#pragma unroll
    for (int c = 0; c < C; ++c) {
        float v = h[i * C + c];
        s += v * aw_s[c];
        d += v * aw_d[c];
    }
    as[i] = s;
    ad[i] = d;
}

// one wave per node: segment softmax + weighted gather, no atomics
template <int C>
__global__ void k_node_agg(const int* __restrict__ rowptr, const int* __restrict__ colidx,
                           const float* __restrict__ as, const float* __restrict__ ad,
                           const float* __restrict__ h, float* __restrict__ agg) {
    int node = blockIdx.x * (blockDim.x >> 5) + (threadIdx.x >> 5);
    if (node >= N_NODES) return;
    int lane = threadIdx.x & 31;
    int beg = rowptr[node], end = rowptr[node + 1];
    float adv = ad[node];

    // pass 1: lane-parallel max of leaky-relu logits, wave-reduce
    float m = -INFINITY;
    for (int e = beg + lane; e < end; e += 32) {
        int s = colidx[e];
        float l = as[s] + adv;
        l = (l > 0.0f) ? l : 0.2f * l;
        m = fmaxf(m, l);
    }
#pragma unroll
    for (int off = 16; off > 0; off >>= 1)
        m = fmaxf(m, __shfl_xor(m, off, 32));

    // pass 2: wave walks edges serially; lanes own C/32 output channels.
    // Each lane recomputes the (uniform) weight; h row load is coalesced 256B.
    constexpr int PER = (C + 31) / 32;
    float acc[PER];
#pragma unroll
    for (int i = 0; i < PER; ++i) acc[i] = 0.0f;
    float wsum = 0.0f;
    for (int e = beg; e < end; ++e) {
        int s = colidx[e];
        float l = as[s] + adv;
        l = (l > 0.0f) ? l : 0.2f * l;
        float w = __expf(l - m);
        wsum += w;
        const float* hs = h + s * C;
#pragma unroll
        for (int i = 0; i < PER; ++i) {
            int c = lane * PER + i;
            if (c < C) acc[i] += hs[c] * w;
        }
    }
    float inv = 1.0f / wsum;   // wsum identical across lanes; deg >= 1 (self loop)
#pragma unroll
    for (int i = 0; i < PER; ++i) {
        int c = lane * PER + i;
        if (c < C) agg[node * C + c] = acc[i] * inv;
    }
}

// ---------------------------------------------------------------------------
extern "C" void kernel_launch(void* const* d_in, const int* in_sizes, int n_in,
                              void* d_out, int out_size, void* d_ws, size_t ws_size,
                              hipStream_t stream) {
    const float* x   = (const float*)d_in[0];
    const int*   ei  = (const int*)d_in[1];   // [2, NE]
    const float* W1  = (const float*)d_in[3];
    const float* a1s = (const float*)d_in[4];
    const float* a1d = (const float*)d_in[5];
    const float* b1  = (const float*)d_in[6];
    const float* W2  = (const float*)d_in[7];
    const float* a2s = (const float*)d_in[8];
    const float* a2d = (const float*)d_in[9];
    const float* b2  = (const float*)d_in[10];
    const float* Wl  = (const float*)d_in[11];
    const float* bl  = (const float*)d_in[12];
    float*       out = (float*)d_out;

    const int* src = ei;
    const int* dst = ei + NE;

    // ---- workspace carving ----
    float* fw   = (float*)d_ws;
    float* h1   = fw;                    // N*16
    float* agg1 = h1   + N_NODES * 16;   // N*16
    float* h2   = agg1 + N_NODES * 16;   // N*64
    float* agg2 = h2   + N_NODES * 64;   // N*64
    float* as   = agg2 + N_NODES * 64;   // N
    float* ad   = as   + N_NODES;        // N
    int* iw     = (int*)(ad + N_NODES);
    int* cnt    = iw;                    // N
    int* rowptr = cnt    + N_NODES;      // N+1
    int* cursor = rowptr + N_NODES + 1;  // N
    int* colidx = cursor + N_NODES;      // NET

    const int GEMM_BLK  = 128;                    // 4 waves / block
    const int GEMM_GRID = (ROW_TILES + 3) / 4;    // 782
    const int NODE_GRID = (N_NODES + 255) / 256;  // 196
    const int EDGE_GRID = (NET + 255) / 256;      // 3321
    const int AGG_GRID  = (N_NODES + 7) / 8;      // 8 waves/block, 1 node/wave

    // ---- CSR by destination (shared by both layers) ----
    k_zero_int<<<NODE_GRID, 256, 0, stream>>>(cnt, N_NODES);
    k_count<<<EDGE_GRID, 256, 0, stream>>>(dst, cnt);
    k_scan<<<1, 1024, 0, stream>>>(cnt, rowptr);
    k_copy_int<<<NODE_GRID, 256, 0, stream>>>(rowptr, cursor, N_NODES);
    k_scatter<<<EDGE_GRID, 256, 0, stream>>>(src, dst, cursor, colidx);

    // ---- layer 1 ----
    k_gemm1<<<GEMM_GRID, GEMM_BLK, 0, stream>>>(x, W1, h1);
    k_alpha<16><<<NODE_GRID, 256, 0, stream>>>(h1, a1s, a1d, as, ad);
    k_node_agg<16><<<AGG_GRID, 256, 0, stream>>>(rowptr, colidx, as, ad, h1, agg1);

    // ---- layer 2 ----
    k_gemm2<<<dim3(GEMM_GRID, 4), GEMM_BLK, 0, stream>>>(agg1, b1, W2, h2);
    k_alpha<64><<<NODE_GRID, 256, 0, stream>>>(h2, a2s, a2d, as, ad);
    k_node_agg<64><<<AGG_GRID, 256, 0, stream>>>(rowptr, colidx, as, ad, h2, agg2);

    // ---- classifier ----
    k_gemm3<<<GEMM_GRID, GEMM_BLK, 0, stream>>>(agg2, b2, Wl, bl, out);
}
